// JepaWorldModel_16003048144932
// MI455X (gfx1250) — compile-verified, matrix operand order checked by reading, not run on hardware
//
#include <hip/hip_runtime.h>

typedef __attribute__((ext_vector_type(8)))  float    v8f;
typedef __attribute__((ext_vector_type(8)))  __bf16   v8bf;
typedef __attribute__((ext_vector_type(16))) __bf16   v16bf;
typedef __attribute__((ext_vector_type(4)))  unsigned v4u;
typedef __attribute__((ext_vector_type(4)))  int      v4i;
typedef __attribute__((ext_vector_type(8)))  int      v8i;

constexpr int BB = 8192;   // batch
constexpr int DD = 4096;   // dim
constexpr int HH = 8192;   // hidden = 2*dim
constexpr int NMASK = 1024;

constexpr int LDS_STRIDE = 40;                 // halves per row (32 data + 4 DW pad)
constexpr int TILE_HALVES = 128 * LDS_STRIDE;  // one buffer, in halves
constexpr unsigned TILE_BYTES = TILE_HALVES * 2;

// ---------- helpers ----------

__device__ __forceinline__ unsigned short f2bf(float x) {
  unsigned u = __float_as_uint(x);
  u += 0x7FFFu + ((u >> 16) & 1u);          // round-to-nearest-even
  return (unsigned short)(u >> 16);
}

// ---------- Tensor Data Mover: 2D tile (32 x 128 bf16) -> LDS ----------
// D# packing per CDNA5 ISA.
// pad_interval=3 (16 DWORDs = one 32-half row), pad_amount=3 (4 DWORDs)
// -> hardware produces the 40-half LDS row stride the fragment loads expect.

__device__ __forceinline__ void tdm_load_tile(unsigned lds_addr, unsigned long long gaddr,
                                              unsigned tdim0, unsigned tdim1,
                                              unsigned kstride) {
  v4u g0;
  g0[0] = 1u;                                              // count=1, user mode
  g0[1] = lds_addr;                                        // LDS byte address
  g0[2] = (unsigned)gaddr;                                 // global addr [31:0]
  g0[3] = (unsigned)((gaddr >> 32) & 0x1FFFFFFu) | (2u << 30); // addr[56:32] | type=2

  const unsigned tile0 = 32u, tile1 = 128u;
  v8i g1;
  g1[0] = (int)((1u << 16) | (1u << 20) | (3u << 22) | (3u << 25)); // 2B elems, pad cfg
  g1[1] = (int)(tdim0 << 16);                              // tensor_dim0 [79:48] lo
  g1[2] = (int)((tdim0 >> 16) | (tdim1 << 16));            // dim0 hi | tensor_dim1 lo
  g1[3] = (int)((tdim1 >> 16) | (tile0 << 16));            // dim1 hi | tile_dim0
  g1[4] = (int)tile1;                                      // tile_dim1 (tile_dim2=0)
  g1[5] = (int)kstride;                                    // tensor_dim0_stride lo32
  g1[6] = 0;                                               // stride hi | dim1_stride lo
  g1[7] = 0;

  v4i z4 = {0, 0, 0, 0};
#if __has_include(<hip/amd_detail/amd_gfx1250_TDM.h>)
  v8i z8 = {0, 0, 0, 0, 0, 0, 0, 0};
  __builtin_amdgcn_tensor_load_to_lds(g0, g1, z4, z4, z8, 0);   // clang-23 arity
#else
  __builtin_amdgcn_tensor_load_to_lds(g0, g1, z4, z4, 0);       // ROCm 7.2 arity
#endif
}

// Load a 16x32 bf16 fragment (A or B operand) from LDS following the CDNA5
// 16-bit operand layout: lanes 0-15 hold K=0..7 / K=16..23, lanes 16-31 hold
// K=8..15 / K=24..31, two contiguous 16B chunks per lane -> ds_load_b128 x2.
__device__ __forceinline__ v16bf frag_from_lds(const unsigned short* s, int lane,
                                               int rbase) {
  int r    = rbase + (lane & 15);
  int koff = (lane >> 4) * 8;
  v8bf lo = *(const v8bf*)(s + r * LDS_STRIDE + koff);
  v8bf hi = *(const v8bf*)(s + r * LDS_STRIDE + 16 + koff);
  return __builtin_shufflevector(lo, hi, 0,1,2,3,4,5,6,7,8,9,10,11,12,13,14,15);
}

// ---------- shared WMMA mainloop (TDM double-buffered) ----------
// 128x128 block tile, 8 waves, 64x32 per wave (4x2 wmma tiles), K-step 32.
// Wave 0 is the TDM producer; TENSORcnt is per-wave so only it waits, then the
// workgroup barrier publishes the tile to the other 7 waves.

template <int KDIM>
__device__ __forceinline__ void wmma_core(const unsigned short* __restrict__ Abase,
                                          const unsigned short* __restrict__ Bbase,
                                          unsigned aRows, unsigned bRows,
                                          const unsigned short* sA,
                                          const unsigned short* sB,
                                          int row0, int col0, int lane, int wave,
                                          v8f acc[4][2]) {
  const unsigned ldsA = (unsigned)(size_t)(const void*)sA;
  const unsigned ldsB = (unsigned)(size_t)(const void*)sB;
  unsigned long long gA = (unsigned long long)(size_t)Abase +
                          (unsigned long long)row0 * KDIM * 2ull;
  unsigned long long gB = (unsigned long long)(size_t)Bbase +
                          (unsigned long long)col0 * KDIM * 2ull;
  const int wm = (wave & 1) * 64;
  const int wn = (wave >> 1) * 32;

  #pragma unroll
  for (int mi = 0; mi < 4; ++mi)
    #pragma unroll
    for (int ni = 0; ni < 2; ++ni)
      #pragma unroll
      for (int e = 0; e < 8; ++e) acc[mi][ni][e] = 0.0f;

  if (wave == 0) {
    tdm_load_tile(ldsA, gA, KDIM, aRows, KDIM);
    tdm_load_tile(ldsB, gB, KDIM, bRows, KDIM);
    __builtin_amdgcn_s_wait_tensorcnt(0);
  }
  __syncthreads();

  const int nk = KDIM / 32;
  for (int kt = 0; kt < nk; ++kt) {
    const int cur = kt & 1, nxt = cur ^ 1;
    if (kt + 1 < nk && wave == 0) {
      const unsigned long long off = (unsigned long long)(kt + 1) * 64ull; // 32 halves
      tdm_load_tile(ldsA + nxt * TILE_BYTES, gA + off, KDIM, aRows, KDIM);
      tdm_load_tile(ldsB + nxt * TILE_BYTES, gB + off, KDIM, bRows, KDIM);
    }
    const unsigned short* pA = sA + cur * TILE_HALVES;
    const unsigned short* pB = sB + cur * TILE_HALVES;
    v16bf af[4], bfr[2];
    #pragma unroll
    for (int mi = 0; mi < 4; ++mi) af[mi]  = frag_from_lds(pA, lane, wm + mi * 16);
    #pragma unroll
    for (int ni = 0; ni < 2; ++ni) bfr[ni] = frag_from_lds(pB, lane, wn + ni * 16);
    #pragma unroll
    for (int mi = 0; mi < 4; ++mi)
      #pragma unroll
      for (int ni = 0; ni < 2; ++ni)
        acc[mi][ni] = __builtin_amdgcn_wmma_f32_16x16x32_bf16(
            false, af[mi], false, bfr[ni], (short)0, acc[mi][ni], false, false);
    if (wave == 0) __builtin_amdgcn_s_wait_tensorcnt(0);
    __syncthreads();
  }
}

// ---------- kernel 0: fp32 -> bf16 weight conversion ----------

__global__ __launch_bounds__(256)
void f32_to_bf16(const float* __restrict__ in, unsigned short* __restrict__ out, int n) {
  int i = blockIdx.x * blockDim.x + threadIdx.x;
  int stride = gridDim.x * blockDim.x;
  for (; i < n; i += stride) out[i] = f2bf(in[i]);
}

// ---------- kernel 1: per-row top-k mask + x_masked (bf16) ----------
// One workgroup (256 threads) per row. Uniform [0,1) floats -> positive ->
// uint bit pattern is order-preserving. 32-pass radix select yields the
// 1024-th smallest key AND the residual rank among equal keys, which
// reproduces stable-argsort tie breaking (lowest index first).

__global__ __launch_bounds__(256)
void mask_kernel(const float* __restrict__ noise, const float* __restrict__ student,
                 unsigned* __restrict__ maskbits, unsigned short* __restrict__ xm) {
  const int row = blockIdx.x;
  const int tid = threadIdx.x;
  const int PER = DD / 256;                 // 16 consecutive elements per thread

  __shared__ unsigned keys[DD];             // 16 KB
  __shared__ int cnt;
  __shared__ int eqpre[256];
  __shared__ unsigned mwords[DD / 32];      // 128 words

  const float* nrow = noise + (size_t)row * DD;
  for (int j = tid; j < DD; j += 256) keys[j] = __float_as_uint(nrow[j]);
  __syncthreads();

  // radix select: find K-th smallest key (K is 1-indexed)
  unsigned prefix = 0;
  int K = NMASK;
  for (int bit = 31; bit >= 0; --bit) {
    if (tid == 0) cnt = 0;
    __syncthreads();
    unsigned pmask = (bit == 31) ? 0u : (0xFFFFFFFFu << (bit + 1));
    int local = 0;
    #pragma unroll
    for (int i = 0; i < PER; ++i) {
      unsigned k = keys[tid * PER + i];
      if (((k ^ prefix) & pmask) == 0 && ((k >> bit) & 1u) == 0u) local++;
    }
    atomicAdd(&cnt, local);
    __syncthreads();
    int c0 = cnt;
    __syncthreads();                        // protect cnt before next reset
    if (K > c0) { K -= c0; prefix |= (1u << bit); }
  }
  const unsigned Tkey = prefix;             // 1024th smallest key
  const int need = K;                       // #equal-keys (by index order) to mask

  // exclusive scan of per-chunk equal counts (index order == chunk order)
  int myeq = 0;
  #pragma unroll
  for (int i = 0; i < PER; ++i) myeq += (keys[tid * PER + i] == Tkey) ? 1 : 0;
  eqpre[tid] = myeq;
  __syncthreads();
  if (tid == 0) {
    int s = 0;
    for (int t = 0; t < 256; ++t) { int v = eqpre[t]; eqpre[t] = s; s += v; }
  }
  if (tid < DD / 32) mwords[tid] = 0u;
  __syncthreads();

  int eqr = eqpre[tid];
  const float* srow = student + (size_t)row * DD;
  unsigned bits16 = 0;
  #pragma unroll
  for (int i = 0; i < PER; ++i) {
    int j = tid * PER + i;
    unsigned k = keys[j];
    bool m;
    if (k < Tkey)       m = true;
    else if (k == Tkey) { m = (eqr < need); eqr++; }
    else                m = false;
    bits16 |= (m ? 1u : 0u) << i;
    xm[(size_t)row * DD + j] = m ? (unsigned short)0 : f2bf(srow[j]);
  }
  atomicOr(&mwords[(tid * PER) >> 5], bits16 << ((tid & 1) * 16));
  __syncthreads();
  if (tid < DD / 32) maskbits[(size_t)row * (DD / 32) + tid] = mwords[tid];
}

// ---------- kernel 2: h = gelu(x_masked @ W1^T + b1), bf16 out ----------

__global__ __launch_bounds__(256)
void gemm1_gelu(const unsigned short* __restrict__ Xm,   // [BB, DD] bf16
                const unsigned short* __restrict__ W1b,  // [HH, DD] bf16
                const float* __restrict__ b1,            // [HH]
                unsigned short* __restrict__ H) {        // [BB, HH] bf16
  const int tid = threadIdx.x, lane = tid & 31, wave = tid >> 5;
  const int row0 = blockIdx.y * 128;       // M
  const int col0 = blockIdx.x * 128;       // N

  __shared__ unsigned short sA[2 * TILE_HALVES];
  __shared__ unsigned short sB[2 * TILE_HALVES];

  v8f acc[4][2];
  wmma_core<DD>(Xm, W1b, BB, HH, sA, sB, row0, col0, lane, wave, acc);

  // epilogue: bias + exact GELU -> bf16
  const int wm = (wave & 1) * 64;
  const int wn = (wave >> 1) * 32;
  const int nloc = lane & 15;
  const int mhi  = (lane >> 4) * 8;
  #pragma unroll
  for (int ni = 0; ni < 2; ++ni) {
    const int col = col0 + wn + ni * 16 + nloc;
    const float bias = b1[col];
    #pragma unroll
    for (int mi = 0; mi < 4; ++mi)
      #pragma unroll
      for (int r = 0; r < 8; ++r) {
        const int row = row0 + wm + mi * 16 + mhi + r;
        const float v = acc[mi][ni][r] + bias;
        const float g = 0.5f * v * (1.0f + erff(v * 0.70710678118654752f));
        H[(size_t)row * HH + col] = f2bf(g);
      }
  }
}

// ---------- kernel 3: pred = h @ W2^T + b2, fused masked-L1 reduction ----------

__global__ __launch_bounds__(256)
void gemm2_loss(const unsigned short* __restrict__ H,    // [BB, HH] bf16
                const unsigned short* __restrict__ W2b,  // [DD, HH] bf16
                const float* __restrict__ b2,            // [DD]
                const float* __restrict__ teacher,       // [BB, DD] f32
                const unsigned* __restrict__ maskbits,   // [BB, DD/32]
                float* __restrict__ acc_out) {
  const int tid = threadIdx.x, lane = tid & 31, wave = tid >> 5;
  const int row0 = blockIdx.y * 128;       // M over BB
  const int col0 = blockIdx.x * 128;       // N over DD

  __shared__ unsigned short sA[2 * TILE_HALVES];
  __shared__ unsigned short sB[2 * TILE_HALVES];
  __shared__ float wred[8];

  v8f acc[4][2];
  wmma_core<HH>(H, W2b, BB, DD, sA, sB, row0, col0, lane, wave, acc);

  // epilogue: bias, |pred - teacher| * mask, block reduction, atomic add
  const int wm = (wave & 1) * 64;
  const int wn = (wave >> 1) * 32;
  const int nloc = lane & 15;
  const int mhi  = (lane >> 4) * 8;
  float lsum = 0.0f;
  #pragma unroll
  for (int ni = 0; ni < 2; ++ni) {
    const int col = col0 + wn + ni * 16 + nloc;
    const float bias = b2[col];
    #pragma unroll
    for (int mi = 0; mi < 4; ++mi)
      #pragma unroll
      for (int r = 0; r < 8; ++r) {
        const int row = row0 + wm + mi * 16 + mhi + r;
        const float v = acc[mi][ni][r] + bias;
        const float t = teacher[(size_t)row * DD + col];
        const unsigned bit =
            (maskbits[(size_t)row * (DD / 32) + (col >> 5)] >> (col & 31)) & 1u;
        if (bit) lsum += fabsf(v - t);
      }
  }
  #pragma unroll
  for (int off = 16; off > 0; off >>= 1) lsum += __shfl_down(lsum, off, 32);
  if (lane == 0) wred[wave] = lsum;
  __syncthreads();
  if (tid == 0) {
    float s = 0.0f;
    for (int i = 0; i < 8; ++i) s += wred[i];
    atomicAdd(acc_out, s);
  }
}

// ---------- tiny kernels: deterministic accumulator init + finalize ----------

__global__ void zero_acc(float* a) { *a = 0.0f; }

__global__ void finalize(const float* a, float* out) {
  *out = *a / (float)(BB * NMASK);          // denominator is static: 8192*1024
}

// ---------- host launch ----------

extern "C" void kernel_launch(void* const* d_in, const int* in_sizes, int n_in,
                              void* d_out, int out_size, void* d_ws, size_t ws_size,
                              hipStream_t stream) {
  const float* student = (const float*)d_in[0];
  const float* teacher = (const float*)d_in[1];
  const float* noise   = (const float*)d_in[2];
  const float* W1      = (const float*)d_in[3];
  const float* b1      = (const float*)d_in[4];
  const float* W2      = (const float*)d_in[5];
  const float* b2      = (const float*)d_in[6];

  char* ws = (char*)d_ws;
  // workspace layout (bytes)
  unsigned short* XM  = (unsigned short*)(ws);                         // 64 MB
  unsigned short* W1B = (unsigned short*)(ws + 67108864ull);           // 64 MB
  unsigned short* W2B = (unsigned short*)(ws + 134217728ull);          // 64 MB
  unsigned short* Hb  = (unsigned short*)(ws + 201326592ull);          // 128 MB
  unsigned*       MB  = (unsigned*)      (ws + 335544320ull);          // 4 MB
  float*          ACC = (float*)         (ws + 339738624ull);          // 4 B

  f32_to_bf16<<<4096, 256, 0, stream>>>(W1, W1B, HH * DD);
  f32_to_bf16<<<4096, 256, 0, stream>>>(W2, W2B, DD * HH);
  mask_kernel<<<BB, 256, 0, stream>>>(noise, student, MB, XM);
  zero_acc<<<1, 1, 0, stream>>>(ACC);

  dim3 g1(HH / 128, BB / 128);   // (64, 64)
  gemm1_gelu<<<g1, 256, 0, stream>>>(XM, W1B, b1, Hb);

  dim3 g2(DD / 128, BB / 128);   // (32, 64)
  gemm2_loss<<<g2, 256, 0, stream>>>(Hb, W2B, b2, teacher, MB, ACC);

  finalize<<<1, 1, 0, stream>>>(ACC, (float*)d_out);
}
